// MHLinearAttention_48661979463968
// MI455X (gfx1250) — matrix operand order repaired and checked by reading, not compiled, not run on hardware
//
#include <hip/hip_runtime.h>
#include <hip/hip_bf16.h>
#include <math.h>

// Problem constants: B=4, N=8192, D=1024, H=16, DK=DV=64
#define SCALE 0.125f   // DK^-0.5

typedef __attribute__((ext_vector_type(16))) __bf16 v16bf;
typedef __attribute__((ext_vector_type(8)))  float  v8f;

union FragBF { unsigned int u[8]; v16bf v; };

__device__ __forceinline__ unsigned short f2bf(float f) {
    unsigned int u = __float_as_uint(f);
    unsigned int r = u + 0x7FFFu + ((u >> 16) & 1u);   // round-to-nearest-even
    return (unsigned short)(r >> 16);
}

// Pack two f32 -> two bf16 (RNE) in one v_perm_b32 splice of the high halves.
__device__ __forceinline__ unsigned int pack2(float a, float b) {
    unsigned int ua = __float_as_uint(a);
    unsigned int ub = __float_as_uint(b);
    ua += 0x7FFFu + ((ua >> 16) & 1u);
    ub += 0x7FFFu + ((ub >> 16) & 1u);
    return __builtin_amdgcn_perm(ub, ua, 0x07060302u);  // {ub.hi16, ua.hi16}
}

// CDNA5 async global->LDS copy (ASYNCcnt-tracked). lds_off = low 32 bits of the
// flat shared pointer (LDS aperture maps addr[31:0] to the LDS offset).
#define ASYNC_LDS_B128(ldsoff, gaddr, IMMOFF)                                  \
    asm volatile("global_load_async_to_lds_b128 %0, %1, off offset:" #IMMOFF   \
                 :: "v"(ldsoff), "v"(gaddr) : "memory")

__device__ __forceinline__ void wait_async_lds() {
    asm volatile("s_wait_asynccnt 0" ::: "memory");
}

// A-matrix 16x32 bf16 fragment (doc layout): lanes 0-15 m=lane K={0..7,16..23},
// lanes 16-31 m=lane-16 K={8..15,24..31}; K-pairs packed per VGPR.
__device__ __forceinline__ v16bf load_frag_a(const unsigned short* s, int row_base, int lda, int k0) {
    int lane = threadIdx.x & 31;
    int m    = lane & 15;
    int koff = (lane >> 4) << 3;                       // 0 or 8
    const unsigned short* p = s + (row_base + m) * lda + k0;
    FragBF f;
#pragma unroll
    for (int j = 0; j < 8; ++j) {
        int k = ((j & 4) ? 16 : 0) + koff + ((j & 3) << 1);
        f.u[j] = *(const unsigned int*)(p + k);
    }
    return f.v;
}

// B-matrix 32x16 bf16 fragment; B staged TRANSPOSED in LDS as (n, k) with K
// contiguous, so lanes 0-15: n=lane K={0..7,16..23}, lanes 16-31: n=lane-16 K={8..15,24..31}.
__device__ __forceinline__ v16bf load_frag_b(const unsigned short* s, int col_base, int ldb, int k0) {
    int lane = threadIdx.x & 31;
    int n    = lane & 15;
    int koff = (lane >> 4) << 3;                       // 0 or 8
    const unsigned short* p = s + (col_base + n) * ldb + k0;
    FragBF f;
#pragma unroll
    for (int j = 0; j < 8; ++j) f.u[j] = *(const unsigned int*)(p + ((j & 4) ? 16 : 0) + koff + ((j & 3) << 1));
    return f.v;
}

__device__ __forceinline__ v8f wmma_bf16(v16bf a, v16bf b, v8f c) {
    return __builtin_amdgcn_wmma_f32_16x16x32_bf16(false, a, false, b, (short)0, c, false, false);
}

// ---------------------------------------------------------------------------
// Kernel 1: fused Q/K/V projection (blockIdx.z selects matrix).
// 128x64 output tile / block, double-buffered LDS, one barrier per K step.
// ---------------------------------------------------------------------------
__global__ __launch_bounds__(256) void proj_kernel(
    const float* __restrict__ qin, const float* __restrict__ kin, const float* __restrict__ vin,
    const float* __restrict__ wq,  const float* __restrict__ wk,  const float* __restrict__ wv,
    unsigned short* __restrict__ qs, float* __restrict__ kp, unsigned short* __restrict__ vp)
{
    __shared__ __align__(16) unsigned short sA[2][128 * 32];   // 16 KB
    __shared__ __align__(16) unsigned short sBT[2][64 * 32];   //  8 KB
    __shared__ float sC[128 * 64];                             // 32 KB

    const int which = blockIdx.z;
    const float* X = (which == 0) ? qin : ((which == 1) ? kin : vin);
    const float* W = (which == 0) ? wq  : ((which == 1) ? wk  : wv);
    const size_t row0 = (size_t)blockIdx.x * 128;      // over B*N = 32768
    const int    col0 = blockIdx.y * 64;               // over 1024 (one head)
    const int tid  = threadIdx.x;
    const int wave = tid >> 5;
    const int lane = tid & 31;

    // per-thread staging geometry
    const int rA  = tid >> 1, kcA = (tid & 1) << 4;        // A tile: 16 elems
    const int kp2 = tid >> 4, nbB = (tid & 15) << 2;       // B tile: k-pair x 4 cols

    auto stage = [&](int buf, int kk) {
        {   // A (128x32) row-major, f32 -> bf16, packed 32-bit LDS stores
            const float* src = X + (row0 + rA) * 1024 + kk + kcA;
            unsigned int* dst = (unsigned int*)(sA[buf] + rA * 32 + kcA);
            __builtin_prefetch(src + 64, 0, 1);            // 2 K-tiles ahead
#pragma unroll
            for (int j = 0; j < 16; j += 4) {
                float4 f = *(const float4*)(src + j);
                dst[(j >> 1) + 0] = pack2(f.x, f.y);
                dst[(j >> 1) + 1] = pack2(f.z, f.w);
            }
        }
        {   // B transposed (n,k): read two adjacent k-rows, pack K-pairs
            const float* s0 = W + (size_t)(kk + 2 * kp2) * 1024 + col0 + nbB;
            const float* s1 = s0 + 1024;
            __builtin_prefetch(s0 + 64 * 1024, 0, 1);
            float4 f0 = *(const float4*)s0;
            float4 f1 = *(const float4*)s1;
            unsigned int* dst = (unsigned int*)sBT[buf];   // uint idx = n*16 + k/2
            dst[(nbB + 0) * 16 + kp2] = pack2(f0.x, f1.x);
            dst[(nbB + 1) * 16 + kp2] = pack2(f0.y, f1.y);
            dst[(nbB + 2) * 16 + kp2] = pack2(f0.z, f1.z);
            dst[(nbB + 3) * 16 + kp2] = pack2(f0.w, f1.w);
        }
    };

    v8f acc[4];
#pragma unroll
    for (int c = 0; c < 4; ++c)
#pragma unroll
        for (int i = 0; i < 8; ++i) acc[c][i] = 0.0f;

    stage(0, 0);
    __syncthreads();
    for (int it = 0; it < 32; ++it) {
        const int cur = it & 1;
        if (it + 1 < 32) stage(cur ^ 1, (it + 1) * 32);
        v16bf a = load_frag_a(sA[cur], wave * 16, 32, 0);
#pragma unroll
        for (int c = 0; c < 4; ++c) {
            v16bf bb = load_frag_b(sBT[cur], c * 16, 32, 0);
            acc[c] = wmma_bf16(a, bb, acc[c]);
        }
        __syncthreads();
    }

    const int nloc = lane & 15;
    const int mhi  = (lane >> 4) << 3;    // C layout: VGPR i -> m = i + 8*(lane>=16)
    if (which == 0) {
#pragma unroll
        for (int c = 0; c < 4; ++c)
#pragma unroll
            for (int i = 0; i < 8; ++i)
                sC[(wave*16 + mhi + i) * 64 + c*16 + nloc] = acc[c][i];
        __syncthreads();
        if (tid < 128) {                   // softmax over this row's 64-wide head
            float* rp = sC + tid * 64;
            float mx = -3.402823466e38f;
            for (int j = 0; j < 64; ++j) { float val = rp[j] * SCALE; rp[j] = val; mx = fmaxf(mx, val); }
            float sum = 0.f;
            for (int j = 0; j < 64; ++j) { float e = __expf(rp[j] - mx); rp[j] = e; sum += e; }
            float inv = 1.0f / sum;
            unsigned short* dst = qs + (row0 + tid) * 1024 + col0;
            for (int j = 0; j < 64; j += 2)
                *(unsigned int*)(dst + j) = pack2(rp[j] * inv, rp[j+1] * inv);
        }
    } else if (which == 1) {
#pragma unroll
        for (int c = 0; c < 4; ++c)
#pragma unroll
            for (int i = 0; i < 8; ++i)
                kp[(row0 + wave*16 + mhi + i) * 1024 + col0 + c*16 + nloc] = acc[c][i];
    } else {
#pragma unroll
        for (int c = 0; c < 4; ++c)
#pragma unroll
            for (int i = 0; i < 8; ++i)
                vp[(row0 + wave*16 + mhi + i) * 1024 + col0 + c*16 + nloc] = f2bf(acc[c][i]);
    }
}

// ---------------------------------------------------------------------------
// Kernel 2: masked column softmax statistics over N (axis=-2 softmax).
// ---------------------------------------------------------------------------
__global__ __launch_bounds__(256) void colstat_kernel(
    const float* __restrict__ kp, const unsigned char* __restrict__ mask,
    float* __restrict__ colmax, float* __restrict__ colden)
{
    const int b = blockIdx.y;
    const int c = blockIdx.x * 256 + threadIdx.x;
    const float* base = kp + (size_t)b * 8192 * 1024 + c;
    const unsigned char* mrow = mask + b * 8192;
    float mx = -3.402823466e38f;
    for (int n = 0; n < 8192; ++n)
        if (mrow[n]) mx = fmaxf(mx, base[(size_t)n * 1024] * SCALE);
    float sum = 0.f;
    for (int n = 0; n < 8192; ++n)
        if (mrow[n]) sum += __expf(base[(size_t)n * 1024] * SCALE - mx);
    colmax[b * 1024 + c] = mx;
    colden[b * 1024 + c] = sum;
}

// ---------------------------------------------------------------------------
// Kernel 3: context[b,h] = softmax_col(kh)^T @ vh  (64x64, K = N = 8192).
// Double-buffered LDS, one barrier per 32-row chunk.
// ---------------------------------------------------------------------------
__global__ __launch_bounds__(256) void ctx_kernel(
    const float* __restrict__ kp, const unsigned short* __restrict__ vp,
    const unsigned char* __restrict__ mask,
    const float* __restrict__ colmax, const float* __restrict__ colden,
    float* __restrict__ ctx)
{
    __shared__ __align__(16) unsigned short sZT[2][64 * 32];   // 8 KB
    __shared__ __align__(16) unsigned short sVT[2][64 * 32];   // 8 KB
    const int bh = blockIdx.x, b = bh >> 4, h = bh & 15;
    const int tid = threadIdx.x, wave = tid >> 5, lane = tid & 31;

    const int nl = tid >> 3;               // 0..31 (n within chunk)
    const int db = (tid & 7) << 3;         // 0..56 (d/e base)

    auto stage = [&](int buf, int n0) {
        int n = n0 + nl;
        size_t gbase = ((size_t)(b * 8192 + n)) * 1024 + h * 64 + db;
        bool mv = mask[b * 8192 + n] != 0;
#pragma unroll
        for (int j = 0; j < 8; ++j) {
            float z = 0.0f;
            if (mv) z = __expf(kp[gbase + j] * SCALE - colmax[b * 1024 + h * 64 + db + j]);
            sZT[buf][(db + j) * 32 + nl] = f2bf(z);
            sVT[buf][(db + j) * 32 + nl] = vp[gbase + j];
        }
    };

    v8f acc[2];
#pragma unroll
    for (int t = 0; t < 2; ++t)
#pragma unroll
        for (int i = 0; i < 8; ++i) acc[t][i] = 0.0f;

    stage(0, 0);
    __syncthreads();
    for (int it = 0; it < 256; ++it) {
        const int cur = it & 1;
        if (it + 1 < 256) stage(cur ^ 1, (it + 1) * 32);
#pragma unroll
        for (int t = 0; t < 2; ++t) {
            int st = wave * 2 + t, mi = st >> 2, ni = st & 3;
            v16bf a  = load_frag_a(sZT[cur], mi * 16, 32, 0);
            v16bf bb = load_frag_b(sVT[cur], ni * 16, 32, 0);
            acc[t] = wmma_bf16(a, bb, acc[t]);
        }
        __syncthreads();
    }
    const int nloc = lane & 15, mhi = (lane >> 4) << 3;
#pragma unroll
    for (int t = 0; t < 2; ++t) {
        int st = wave * 2 + t, mi = st >> 2, ni = st & 3;
#pragma unroll
        for (int i = 0; i < 8; ++i) {
            int d = mi * 16 + mhi + i, e = ni * 16 + nloc;
            float den = colden[b * 1024 + h * 64 + d];
            ctx[((size_t)bh * 64 + d) * 64 + e] = acc[t][i] / den;
        }
    }
}

// ---------------------------------------------------------------------------
// Kernel 4: attn[b,h] = qs @ context  (M=8192 rows, N=64, K=64), bf16 out.
// Q tile staged with CDNA5 async global->LDS (bf16 passthrough, no VGPRs).
// ---------------------------------------------------------------------------
__global__ __launch_bounds__(256) void attn_kernel(
    const unsigned short* __restrict__ qs, const float* __restrict__ ctx,
    unsigned short* __restrict__ attn)
{
    __shared__ __align__(16) unsigned short sQ[128 * 64];
    __shared__ __align__(16) unsigned short sCT[64 * 64];
    const int bh = blockIdx.y, b = bh >> 4, h = bh & 15;
    const size_t row0 = (size_t)blockIdx.x * 128;
    const int tid = threadIdx.x, wave = tid >> 5, lane = tid & 31;

    {   // stage Q tile: 64B per thread via 4 async b128 copies
        int r = tid >> 1, cc = (tid & 1) << 5;
        unsigned lds0 = (unsigned)(size_t)(sQ + r * 64 + cc);
        unsigned long long g0 =
            (unsigned long long)(size_t)(qs + ((size_t)(b * 8192) + row0 + r) * 1024 + h * 64 + cc);
        ASYNC_LDS_B128(lds0, g0, 0);
        ASYNC_LDS_B128(lds0, g0, 16);
        ASYNC_LDS_B128(lds0, g0, 32);
        ASYNC_LDS_B128(lds0, g0, 48);
    }
    {   // stage context transposed (e,d), f32 -> bf16
        int d = tid >> 2, eb = (tid & 3) << 4;
        const float* src = ctx + ((size_t)bh * 64 + d) * 64 + eb;
#pragma unroll
        for (int j = 0; j < 16; ++j) sCT[(eb + j) * 64 + d] = f2bf(src[j]);
    }
    wait_async_lds();
    __syncthreads();

    v8f acc[4];
#pragma unroll
    for (int c = 0; c < 4; ++c)
#pragma unroll
        for (int i = 0; i < 8; ++i) acc[c][i] = 0.0f;
#pragma unroll
    for (int kk = 0; kk < 64; kk += 32) {
        v16bf a = load_frag_a(sQ, wave * 16, 64, kk);
#pragma unroll
        for (int c = 0; c < 4; ++c) {
            v16bf bb = load_frag_b(sCT, c * 16, 64, kk);
            acc[c] = wmma_bf16(a, bb, acc[c]);
        }
    }
    const int nloc = lane & 15, mhi = (lane >> 4) << 3;
#pragma unroll
    for (int c = 0; c < 4; ++c)
#pragma unroll
        for (int i = 0; i < 8; ++i) {
            size_t n = row0 + wave * 16 + mhi + i;
            attn[((size_t)(b * 8192) + n) * 1024 + h * 64 + c * 16 + nloc] = f2bf(acc[c][i]);
        }
}

// ---------------------------------------------------------------------------
// Kernel 5: out = attn @ w_fc + residual   (32768x1024, K=1024)
// Double-buffered; A tile (already bf16) staged via async global->LDS.
// ---------------------------------------------------------------------------
__global__ __launch_bounds__(256) void fc_kernel(
    const unsigned short* __restrict__ attn, const float* __restrict__ wfc,
    const float* __restrict__ residual, float* __restrict__ out)
{
    __shared__ __align__(16) unsigned short sA[2][128 * 32];   // 16 KB
    __shared__ __align__(16) unsigned short sBT[2][64 * 32];   //  8 KB
    const size_t row0 = (size_t)blockIdx.x * 128;
    const int    col0 = blockIdx.y * 64;
    const int tid = threadIdx.x, wave = tid >> 5, lane = tid & 31;

    const int rA  = tid >> 1, kcA = (tid & 1) << 4;
    const int kp2 = tid >> 4, nbB = (tid & 15) << 2;

    auto stage = [&](int buf, int kk) {
        {   // A: 32B per thread, async global->LDS (bf16 passthrough)
            unsigned ldsA = (unsigned)(size_t)(sA[buf] + rA * 32 + kcA);
            unsigned long long g0 =
                (unsigned long long)(size_t)(attn + (row0 + rA) * 1024 + kk + kcA);
            ASYNC_LDS_B128(ldsA, g0, 0);
            ASYNC_LDS_B128(ldsA, g0, 16);
        }
        {   // w_fc transposed, f32 -> bf16, packed K-pairs
            const float* s0 = wfc + (size_t)(kk + 2 * kp2) * 1024 + col0 + nbB;
            const float* s1 = s0 + 1024;
            __builtin_prefetch(s0 + 64 * 1024, 0, 1);
            float4 f0 = *(const float4*)s0;
            float4 f1 = *(const float4*)s1;
            unsigned int* dst = (unsigned int*)sBT[buf];
            dst[(nbB + 0) * 16 + kp2] = pack2(f0.x, f1.x);
            dst[(nbB + 1) * 16 + kp2] = pack2(f0.y, f1.y);
            dst[(nbB + 2) * 16 + kp2] = pack2(f0.z, f1.z);
            dst[(nbB + 3) * 16 + kp2] = pack2(f0.w, f1.w);
        }
    };

    v8f acc[4];
#pragma unroll
    for (int c = 0; c < 4; ++c)
#pragma unroll
        for (int i = 0; i < 8; ++i) acc[c][i] = 0.0f;

    stage(0, 0);
    wait_async_lds();
    __syncthreads();
    for (int it = 0; it < 32; ++it) {
        const int cur = it & 1;
        if (it + 1 < 32) stage(cur ^ 1, (it + 1) * 32);
        v16bf a = load_frag_a(sA[cur], wave * 16, 32, 0);
#pragma unroll
        for (int c = 0; c < 4; ++c) {
            v16bf bb = load_frag_b(sBT[cur], c * 16, 32, 0);
            acc[c] = wmma_bf16(a, bb, acc[c]);
        }
        wait_async_lds();
        __syncthreads();
    }
    const int nloc = lane & 15, mhi = (lane >> 4) << 3;
#pragma unroll
    for (int c = 0; c < 4; ++c)
#pragma unroll
        for (int i = 0; i < 8; ++i) {
            size_t idx = (row0 + wave * 16 + mhi + i) * 1024 + col0 + c * 16 + nloc;
            out[idx] = acc[c][i] + residual[idx];
        }
}

// ---------------------------------------------------------------------------
// Kernel 6: in-place LayerNorm over D=1024 per row.
// ---------------------------------------------------------------------------
__global__ __launch_bounds__(256) void ln_kernel(
    float* __restrict__ out, const float* __restrict__ gamma, const float* __restrict__ beta)
{
    __shared__ float red[256];
    const size_t row = blockIdx.x;
    const int tid = threadIdx.x;
    float x[4];
    float s = 0.f;
#pragma unroll
    for (int j = 0; j < 4; ++j) { x[j] = out[row * 1024 + j * 256 + tid]; s += x[j]; }
    red[tid] = s; __syncthreads();
    for (int o = 128; o > 0; o >>= 1) { if (tid < o) red[tid] += red[tid + o]; __syncthreads(); }
    float mu = red[0] * (1.0f / 1024.0f);
    __syncthreads();
    float vs = 0.f;
#pragma unroll
    for (int j = 0; j < 4; ++j) { float d = x[j] - mu; vs += d * d; }
    red[tid] = vs; __syncthreads();
    for (int o = 128; o > 0; o >>= 1) { if (tid < o) red[tid] += red[tid + o]; __syncthreads(); }
    float inv = rsqrtf(red[0] * (1.0f / 1024.0f) + 1e-6f);
#pragma unroll
    for (int j = 0; j < 4; ++j) {
        int c = j * 256 + tid;
        out[row * 1024 + c] = (x[j] - mu) * inv * gamma[c] + beta[c];
    }
}

// ---------------------------------------------------------------------------
extern "C" void kernel_launch(void* const* d_in, const int* in_sizes, int n_in,
                              void* d_out, int out_size, void* d_ws, size_t ws_size,
                              hipStream_t stream)
{
    (void)in_sizes; (void)n_in; (void)out_size; (void)ws_size;
    const float*         q     = (const float*)d_in[0];
    const float*         k     = (const float*)d_in[1];
    const float*         v     = (const float*)d_in[2];
    const unsigned char* mask  = (const unsigned char*)d_in[3];
    const float*         w_qs  = (const float*)d_in[4];
    const float*         w_ks  = (const float*)d_in[5];
    const float*         w_vs  = (const float*)d_in[6];
    const float*         w_fc  = (const float*)d_in[7];
    const float*         gamma = (const float*)d_in[8];
    const float*         beta  = (const float*)d_in[9];
    float* out = (float*)d_out;

    char* ws = (char*)d_ws;
    size_t off = 0;
    float*          kp     = (float*)(ws + off);          off += (size_t)4 * 8192 * 1024 * 4;
    unsigned short* vp     = (unsigned short*)(ws + off); off += (size_t)4 * 8192 * 1024 * 2;
    unsigned short* qs     = (unsigned short*)(ws + off); off += (size_t)4 * 8192 * 1024 * 2;
    unsigned short* attn   = (unsigned short*)(ws + off); off += (size_t)4 * 8192 * 1024 * 2;
    float*          colmax = (float*)(ws + off);          off += (size_t)4 * 1024 * 4;
    float*          colden = (float*)(ws + off);          off += (size_t)4 * 1024 * 4;
    float*          ctx    = (float*)(ws + off);          off += (size_t)64 * 64 * 64 * 4;

    proj_kernel   <<<dim3(256, 16, 3), 256, 0, stream>>>(q, k, v, w_qs, w_ks, w_vs, qs, kp, vp);
    colstat_kernel<<<dim3(4, 4),       256, 0, stream>>>(kp, mask, colmax, colden);
    ctx_kernel    <<<dim3(64),         256, 0, stream>>>(kp, vp, mask, colmax, colden, ctx);
    attn_kernel   <<<dim3(64, 64),     256, 0, stream>>>(qs, ctx, attn);
    fc_kernel     <<<dim3(256, 16),    256, 0, stream>>>(attn, w_fc, q, out);
    ln_kernel     <<<dim3(32768),      256, 0, stream>>>(out, gamma, beta);
}
